// hierarch_sep_loss_70308614635973
// MI455X (gfx1250) — compile-verified, matrix operand order checked by reading, not run on hardware
//
#include <hip/hip_runtime.h>
#include <hip/hip_bf16.h>

typedef __attribute__((ext_vector_type(16))) _Float16 v16h;
typedef __attribute__((ext_vector_type(8)))  float    v8f;
typedef __attribute__((ext_vector_type(4)))  float    v4f;

#define NB    100
#define MODS  2
#define EPSF  1e-10f
#define CENW  0.5f

// acc layout (8 floats in d_ws): [mod*4 + 0]=Su  [1]=Sc  [2]=cntU  [3]=cntC
__global__ void hsl_main_kernel(const float* __restrict__ preds,
                                const int*   __restrict__ labs,
                                const int*   __restrict__ cen,
                                float*       __restrict__ acc,
                                int N)
{
    const int lane      = threadIdx.x & 31;
    const int rowInTile = lane & 15;
    const int hi        = lane >> 4;        // 0: K-low half, 1: K-high half
    const int kbase     = hi * 8;           // A-matrix 16-bit layout K base

    const int waveId   = blockIdx.x * (blockDim.x >> 5) + (threadIdx.x >> 5);
    const int nWaves   = gridDim.x * (blockDim.x >> 5);
    const int tilesPM  = N >> 4;            // 16-row tiles per mod
    const int nTiles   = MODS * tilesPM;

    // B matrix: all ones (f16) -> D[m][n] = row-sum of A for every n
    v16h bOnes;
#pragma unroll
    for (int i = 0; i < 16; ++i) bOnes[i] = (_Float16)1.0f;

    float la[8];
#pragma unroll
    for (int i = 0; i < 8; ++i) la[i] = 0.0f;

    for (int tile = waveId; tile < nTiles; tile += nWaves) {
        const int mod  = tile / tilesPM;
        const int tIn  = tile - mod * tilesPM;
        const int row0 = tIn << 4;                       // first row (within mod)
        const int myRow = row0 + rowInTile;              // this lane's row (within mod)
        const long long myRowG = (long long)mod * N + myRow;
        const float* __restrict__ prow = preds + myRowG * NB;

        const int t1 = labs[myRow] + 1;                  // labs0 threshold (mod 0)

        // prefetch next tile's data for this lane (speculative)
        __builtin_prefetch(prow + (long long)nWaves * 16 * NB, 0, 3);

        // --- masked row sums over K = 0..95 via 3x v_wmma_f32_16x16x32_f16 ---
        v8f c = {};
#pragma unroll
        for (int kb = 0; kb < 96; kb += 32) {
            // two contiguous 8-float runs per lane (A-matrix 16-bit layout)
            const int o0 = kb + kbase;        // K = o0..o0+7
            const int o1 = kb + kbase + 16;   // K = o1..o1+7
            v4f r0a = *(const v4f*)(prow + o0);
            v4f r0b = *(const v4f*)(prow + o0 + 4);
            v4f r1a = *(const v4f*)(prow + o1);
            v4f r1b = *(const v4f*)(prow + o1 + 4);
            v16h a;
#pragma unroll
            for (int e = 0; e < 8; ++e) {
                const float f0 = (e < 4) ? r0a[e] : r0b[e - 4];
                const float f1 = (e < 4) ? r1a[e] : r1b[e - 4];
                a[e]     = (_Float16)(((o0 + e) >= t1) ? f0 : 0.0f);
                a[8 + e] = (_Float16)(((o1 + e) >= t1) ? f1 : 0.0f);
            }
            c = __builtin_amdgcn_wmma_f32_16x16x32_f16(
                    /*neg_a=*/false, a, /*neg_b=*/false, bOnes,
                    /*c_mod=*/(short)0, c, /*reuse_a=*/false, /*reuse_b=*/false);
        }

        // --- censored loss: lanes 0 and 16 own rows 0-7 / 8-15 of D ---
        float cenContrib = 0.0f;
        if (rowInTile == 0) {
#pragma unroll
            for (int r = 0; r < 8; ++r) {
                const int rr = row0 + hi * 8 + r;        // row within mod
                const long long rrG = (long long)mod * N + rr;
                const int   cenR = cen[rrG];
                const int   t1R  = labs[rr] + 1;
                const float* pr = preds + rrG * NB;
                float tail = 0.0f;                        // columns 96..99
#pragma unroll
                for (int k = 96; k < NB; ++k)
                    tail += (k >= t1R) ? pr[k] : 0.0f;
                const float s = c[r] + tail;
                cenContrib += (cenR == 1) ? -logf(s + EPSF) : 0.0f;
            }
        }

        // --- uncensored loss + counts: lanes 0..15 own one row each ---
        const int myLab = labs[myRowG];
        const int myCen = cen[myRowG];
        const float g   = prow[myLab];
        const float uncenContrib = (hi == 0 && myCen == 0) ? -logf(g + EPSF) : 0.0f;
        const float cntU = (hi == 0 && myCen == 0) ? 1.0f : 0.0f;
        const float cntC = (hi == 0 && myCen == 1) ? 1.0f : 0.0f;

        la[mod * 4 + 0] += uncenContrib;
        la[mod * 4 + 1] += cenContrib;
        la[mod * 4 + 2] += cntU;
        la[mod * 4 + 3] += cntC;
    }

    // --- wave reduction (wave32) + one atomic per accumulator per wave ---
#pragma unroll
    for (int i = 0; i < 8; ++i) {
        float v = la[i];
#pragma unroll
        for (int off = 16; off > 0; off >>= 1)
            v += __shfl_xor(v, off, 32);
        if (lane == 0 && v != 0.0f)
            atomicAdd(&acc[i], v);
    }

    // --- remainder rows (N % 16), scalar path; empty for N = 500000 ---
    const int rem = N - (tilesPM << 4);
    if (rem > 0) {
        const int tid = blockIdx.x * blockDim.x + threadIdx.x;
        const int nT  = gridDim.x * blockDim.x;
        for (int j = tid; j < MODS * rem; j += nT) {
            const int mod = j / rem;
            const int i   = (tilesPM << 4) + (j - mod * rem);
            const long long iG = (long long)mod * N + i;
            const float* pr = preds + iG * NB;
            const int cenR = cen[iG];
            if (cenR == 0) {
                atomicAdd(&acc[mod * 4 + 0], -logf(pr[labs[iG]] + EPSF));
                atomicAdd(&acc[mod * 4 + 2], 1.0f);
            } else {
                const int t1R = labs[i] + 1;
                float s = 0.0f;
                for (int k = t1R; k < NB; ++k) s += pr[k];
                atomicAdd(&acc[mod * 4 + 1], -logf(s + EPSF));
                atomicAdd(&acc[mod * 4 + 3], 1.0f);
            }
        }
    }
}

__global__ void hsl_finalize_kernel(const float* __restrict__ acc,
                                    float* __restrict__ out, int N)
{
    if (threadIdx.x == 0 && blockIdx.x == 0) {
        const float n = (float)N;
        float total = 0.0f;
#pragma unroll
        for (int m = 0; m < MODS; ++m) {
            const float Su = acc[m * 4 + 0];
            const float Sc = acc[m * 4 + 1];
            const float cu = acc[m * 4 + 2];
            const float cc = acc[m * 4 + 3];
            const float uncenLoss = Su / fmaxf(1.0f, cu);
            const float cenLoss   = Sc / fmaxf(1.0f, cc);
            total += (cu / n) * uncenLoss + (cc / n) * cenLoss * CENW;
        }
        out[0] = total;
    }
}

extern "C" void kernel_launch(void* const* d_in, const int* in_sizes, int n_in,
                              void* d_out, int out_size, void* d_ws, size_t ws_size,
                              hipStream_t stream) {
    const float* preds = (const float*)d_in[0];   // [2, N, 100] f32
    const int*   labs  = (const int*)d_in[1];     // [2, N] int32
    const int*   cenin = (const int*)d_in[2];     // [2, N] int32
    float* acc = (float*)d_ws;                    // 8 f32 accumulators
    float* out = (float*)d_out;                   // scalar f32

    const int N = in_sizes[1] / MODS;             // 500000

    hipMemsetAsync(acc, 0, 8 * sizeof(float), stream);

    const int block = 256;                        // 8 wave32s
    const int grid  = 512;                        // 4096 waves, grid-stride tiles
    hsl_main_kernel<<<grid, block, 0, stream>>>(preds, labs, cenin, acc, N);
    hsl_finalize_kernel<<<1, 32, 0, stream>>>(acc, out, N);
}
